// Cell_37099927503006
// MI455X (gfx1250) — compile-verified, matrix operand order checked by reading, not run on hardware
//
#include <hip/hip_runtime.h>

// ---------------------------------------------------------------------------
// DARTS mixed cell on MI455X (gfx1250, wave32).
// Pointwise 1x1 convs -> bf16 WMMA GEMMs (v_wmma_f32_16x16x32_bf16),
// 64-pixel M-tiles, double-buffered LDS A-stage, B-fragment reuse x4,
// A-fragment loads batched ahead of back-to-back WMMA issue.
// Depthwise convs / pools / gates -> VALU kernels.
// ---------------------------------------------------------------------------

#define HW   1024            // 32x32 spatial
#define NB   32              // batch
#define CCH  128             // cell channel count
#define BN_SCALE 0.9999950000374997f   // 1/sqrt(1+1e-5)

#define LDSPITCH 40          // padded row stride (ushorts) for 32-ch A rows

typedef __attribute__((ext_vector_type(16))) __bf16 v16bf;
typedef __attribute__((ext_vector_type(8)))  float  v8f;

__device__ __forceinline__ unsigned short f2bf(float f) {
    union { float f; unsigned int u; } c; c.f = f;
    unsigned int u = c.u;
    u += 0x7FFFu + ((u >> 16) & 1u);      // round-to-nearest-even
    return (unsigned short)(u >> 16);
}

// ---------------------------------------------------------------------------
// zero / bf16-convert helpers
// ---------------------------------------------------------------------------
__global__ __launch_bounds__(256) void k_zero(float* p, int n) {
    int i = blockIdx.x * 256 + threadIdx.x;
    if (i < n) p[i] = 0.0f;
}

__global__ __launch_bounds__(256) void k_cvt(const float* __restrict__ s,
                                             unsigned short* __restrict__ d, int n) {
    int i = blockIdx.x * 256 + threadIdx.x;
    if (i < n) d[i] = f2bf(s[i]);
}

// ---------------------------------------------------------------------------
// gates: per (m,b) top-2 masked softmax over 8 ops -> p[m][b][8]
// ---------------------------------------------------------------------------
__global__ __launch_bounds__(256) void k_gate(const float* __restrict__ g,
                                              float* __restrict__ p) {
    int i = blockIdx.x * 256 + threadIdx.x;      // 0..447  (14*32)
    if (i >= 14 * 32) return;
    const float* gi = g + i * 8;
    float v[8];
    for (int o = 0; o < 8; ++o) v[o] = gi[o];
    int i1 = 0;
    for (int o = 1; o < 8; ++o) if (v[o] > v[i1]) i1 = o;
    int i2 = (i1 == 0) ? 1 : 0;
    for (int o = 0; o < 8; ++o) if (o != i1 && v[o] > v[i2]) i2 = o;
    float m  = fmaxf(v[i1], v[i2]);
    float e1 = __expf(v[i1] - m), e2 = __expf(v[i2] - m);
    float inv = 1.0f / (e1 + e2);
    float out[8];
    for (int o = 0; o < 8; ++o) out[o] = 0.0f;
    out[i1] = e1 * inv; out[i2] = e2 * inv;
    float* pi = p + i * 8;
    for (int o = 0; o < 8; ++o) pi[o] = out[o];
}

// ---------------------------------------------------------------------------
// A-stage: 64 pixels x 32 channels, relu + f32->bf16, fully straight-line.
// Thread t: rows r = t&63 (64 consecutive floats per channel -> coalesced),
// channels c = q*4 + (t>>6), q = 0..7.
// ---------------------------------------------------------------------------
__device__ __forceinline__ void stageA(const float* __restrict__ xb, int k0,
                                       unsigned short* __restrict__ buf,
                                       int tid, int applyRelu) {
    const int r  = tid & 63;
    const int cb = tid >> 6;            // 0..3
    float v[8];
#pragma unroll
    for (int q = 0; q < 8; ++q)
        v[q] = xb[(long)(k0 + q * 4 + cb) * HW + r];
#pragma unroll
    for (int q = 0; q < 8; ++q) {
        float f = applyRelu ? fmaxf(v[q], 0.0f) : v[q];
        buf[r * LDSPITCH + q * 4 + cb] = f2bf(f);
    }
}

// ---------------------------------------------------------------------------
// WMMA GEMM: out[b,o,p] (+)= scale(b) * BN * sum_k w[o,k] * (relu?)x[b,k,p]
//   x : f32, channel stride HW, batch stride xBS (may point into d_out)
//   wb: bf16 bits, row-major [128][Cin]
//   block = 256 threads = 8 waves; tile = 64 pixels x 128 outputs
//   each wave: 4 M-subtiles -> 4 wmma per 32-wide K-step, B fragment reused x4
// ---------------------------------------------------------------------------
__global__ __launch_bounds__(256) void k_gemm(const float* __restrict__ x, long xBS,
                                              const unsigned short* __restrict__ wb,
                                              float* __restrict__ dst, long dBS,
                                              const float* __restrict__ gate, // stride-8 per b, or null
                                              int Cin, int applyRelu, int accumulate) {
    __shared__ __align__(16) unsigned short Alds[2][64 * LDSPITCH];  // 10 KB

    const int tile = blockIdx.x;               // 0..511
    const int b    = (tile * 64) / HW;
    const int p0   = (tile * 64) % HW;
    const int tid  = threadIdx.x;
    const int lane = tid & 31;
    const int wave = tid >> 5;                 // 0..7 -> output-channel base 16*wave

    const float* xb = x + (long)b * xBS + p0;

    const int r     = lane & 15;               // row within 16-pixel subtile
    const int khalf = lane >> 4;               // 0 or 1 (K-half selector)
    const int ocol  = wave * 16 + (lane & 15); // output channel for B/C

    v8f acc[4];
#pragma unroll
    for (int s = 0; s < 4; ++s) acc[s] = (v8f){};

    // prologue: stage first K-slab
    stageA(xb, 0, Alds[0], tid, applyRelu);

    int ibuf = 0;
    for (int k0 = 0; k0 < Cin; k0 += 32, ibuf ^= 1) {
        __syncthreads();
        if (k0 + 32 < Cin) {
            stageA(xb, k0 + 32, Alds[ibuf ^ 1], tid, applyRelu);
            __builtin_prefetch((const void*)(xb + (long)(k0 + 64) * HW), 0, 0);
        }

        // ---- B fragment (32x16 bf16) straight from global weights ----
        union Frag { unsigned int u[8]; v16bf v; };
        Frag B;
        const unsigned int* wl = (const unsigned int*)(wb + (long)ocol * Cin + k0);
#pragma unroll
        for (int q = 0; q < 8; ++q) {
            int kk = ((q < 4) ? (2 * q) : (16 + 2 * (q - 4))) + (khalf << 3);
            B.u[q] = wl[kk >> 1];
        }

        // ---- load ALL 4 A fragments first (one ds clause, one wait) ----
        const unsigned int* al = (const unsigned int*)(Alds[ibuf]);
        Frag A[4];
#pragma unroll
        for (int s = 0; s < 4; ++s) {
            const int row20 = (s * 16 + r) * (LDSPITCH / 2);
#pragma unroll
            for (int q = 0; q < 8; ++q) {
                int base = (q < 4) ? q : (8 + (q - 4));
                A[s].u[q] = al[row20 + base + (khalf << 2)];
            }
        }
        // ---- then 4 WMMAs back-to-back, reusing B ----
#pragma unroll
        for (int s = 0; s < 4; ++s)
            acc[s] = __builtin_amdgcn_wmma_f32_16x16x32_bf16(
                         false, A[s].v, false, B.v, (short)0, acc[s], false, false);
    }

    // ---- store: lane owns channel `ocol`; subtile s -> pixels p0+s*16+mg*8.. ----
    float sc = BN_SCALE;
    if (gate) sc *= gate[b * 8];
    const int mg = lane >> 4;
#pragma unroll
    for (int s = 0; s < 4; ++s) {
        float* dp = dst + (long)b * dBS + (long)ocol * HW + p0 + s * 16 + mg * 8;
        if (accumulate) {
            v8f old = *(v8f*)dp;
#pragma unroll
            for (int q = 0; q < 8; ++q) old[q] += sc * acc[s][q];
            *(v8f*)dp = old;
        } else {
            v8f o2;
#pragma unroll
            for (int q = 0; q < 8; ++q) o2[q] = sc * acc[s][q];
            *(v8f*)dp = o2;
        }
    }
}

// ---------------------------------------------------------------------------
// depthwise conv with fused input relu: y[b,c,p] = sum relu(x) * w[c,k,k]
// y is a contiguous [32][128][1024] tmp buffer
// ---------------------------------------------------------------------------
__global__ __launch_bounds__(256) void k_dw(const float* __restrict__ x, long xBS,
                                            const float* __restrict__ w,
                                            float* __restrict__ y,
                                            int k, int pad, int dil) {
    int idx = blockIdx.x * 256 + threadIdx.x;            // 0 .. 32*128*1024-1
    if (idx >= NB * CCH * HW) return;
    int pix = idx & (HW - 1);
    int c   = (idx >> 10) & (CCH - 1);
    int b   = idx >> 17;
    int h = pix >> 5, ww = pix & 31;
    const float* xc = x + (long)b * xBS + (long)c * HW;
    const float* wc = w + c * k * k;
    float s = 0.0f;
    for (int i = 0; i < k; ++i) {
        int hh = h + i * dil - pad;
        if (hh < 0 || hh >= 32) continue;
        for (int j = 0; j < k; ++j) {
            int w2 = ww + j * dil - pad;
            if (w2 < 0 || w2 >= 32) continue;
            s += fmaxf(xc[hh * 32 + w2], 0.0f) * wc[i * k + j];
        }
    }
    y[idx] = s;
}

// ---------------------------------------------------------------------------
// pool + skip: dst += g1*BN*max3(x) + g2*BN*avg3(x) + g3*x
// ---------------------------------------------------------------------------
__global__ __launch_bounds__(256) void k_pool(const float* __restrict__ x, long xBS,
                                              float* __restrict__ dst, long dBS,
                                              const float* __restrict__ pg) {
    int idx = blockIdx.x * 256 + threadIdx.x;
    if (idx >= NB * CCH * HW) return;
    int pix = idx & (HW - 1);
    int c   = (idx >> 10) & (CCH - 1);
    int b   = idx >> 17;
    int h = pix >> 5, ww = pix & 31;
    const float* xc = x + (long)b * xBS + (long)c * HW;
    float mx = -3.402823466e38f, sum = 0.0f;
    int cnt = 0;
    for (int di = -1; di <= 1; ++di) {
        int hh = h + di;
        if (hh < 0 || hh >= 32) continue;
        for (int dj = -1; dj <= 1; ++dj) {
            int w2 = ww + dj;
            if (w2 < 0 || w2 >= 32) continue;
            float v = xc[hh * 32 + w2];
            mx = fmaxf(mx, v);
            sum += v;
            ++cnt;
        }
    }
    float ctr = xc[h * 32 + ww];
    float g1 = pg[b * 8 + 1], g2 = pg[b * 8 + 2], g3 = pg[b * 8 + 3];
    dst[(long)b * dBS + (long)c * HW + pix] +=
        g1 * BN_SCALE * mx + g2 * BN_SCALE * (sum / (float)cnt) + g3 * ctr;
}

// ---------------------------------------------------------------------------
// host launcher
// ---------------------------------------------------------------------------
extern "C" void kernel_launch(void* const* d_in, const int* in_sizes, int n_in,
                              void* d_out, int out_size, void* d_ws, size_t ws_size,
                              hipStream_t stream) {
    (void)in_sizes; (void)n_in; (void)out_size; (void)ws_size;

    const float* s0       = (const float*)d_in[0];
    const float* s1       = (const float*)d_in[1];
    const float* gates    = (const float*)d_in[2];
    const float* pre0_w   = (const float*)d_in[3];
    const float* pre1_w   = (const float*)d_in[4];
    const float* sep3_dw1 = (const float*)d_in[5];
    const float* sep3_pw1 = (const float*)d_in[6];
    const float* sep3_dw2 = (const float*)d_in[7];
    const float* sep3_pw2 = (const float*)d_in[8];
    const float* sep5_dw1 = (const float*)d_in[9];
    const float* sep5_pw1 = (const float*)d_in[10];
    const float* sep5_dw2 = (const float*)d_in[11];
    const float* sep5_pw2 = (const float*)d_in[12];
    const float* dil3_dw  = (const float*)d_in[13];
    const float* dil3_pw  = (const float*)d_in[14];
    const float* dil5_dw  = (const float*)d_in[15];
    const float* dil5_pw  = (const float*)d_in[16];
    float* out = (float*)d_out;

    // workspace layout (~70 MB):
    //   4 x 16MB f32 buffers | 16KB gates | ~3MB bf16 weights
    char* ws = (char*)d_ws;
    const size_t SBYTES = (size_t)NB * CCH * HW * sizeof(float);   // 16 MiB
    float* st0  = (float*)(ws);
    float* st1  = (float*)(ws + SBYTES);
    float* tmp1 = (float*)(ws + 2 * SBYTES);
    float* tmp2 = (float*)(ws + 3 * SBYTES);
    float* pG   = (float*)(ws + 4 * SBYTES);
    unsigned short* wpre0 = (unsigned short*)(ws + 4 * SBYTES + 16384);
    unsigned short* wpre1 = wpre0 + 65536;
    unsigned short* w3p1  = wpre1 + 65536;
    unsigned short* w3p2  = w3p1 + 14 * 16384;
    unsigned short* w5p1  = w3p2 + 14 * 16384;
    unsigned short* w5p2  = w5p1 + 14 * 16384;
    unsigned short* wd3   = w5p2 + 14 * 16384;
    unsigned short* wd5   = wd3  + 14 * 16384;

    const int NEL   = NB * CCH * HW;          // 4,194,304 elements per state
    const int GDW   = NEL / 256;              // 16384 blocks
    const int GOUT  = NB * 512 * HW / 256;    // zero grid for d_out
    const int GGEMM = (NB * HW) / 64;         // 512 blocks (64-pixel M-tiles)

    // 1) zero output (states 2..5 accumulate in place)
    k_zero<<<GOUT, 256, 0, stream>>>(out, NB * 512 * HW);
    // 2) gate softmax
    k_gate<<<2, 256, 0, stream>>>(gates, pG);
    // 3) weights -> bf16
    k_cvt<<<(65536 + 255) / 256, 256, 0, stream>>>(pre0_w, wpre0, 65536);
    k_cvt<<<(65536 + 255) / 256, 256, 0, stream>>>(pre1_w, wpre1, 65536);
    const int NPW = 14 * 16384;
    k_cvt<<<(NPW + 255) / 256, 256, 0, stream>>>(sep3_pw1, w3p1, NPW);
    k_cvt<<<(NPW + 255) / 256, 256, 0, stream>>>(sep3_pw2, w3p2, NPW);
    k_cvt<<<(NPW + 255) / 256, 256, 0, stream>>>(sep5_pw1, w5p1, NPW);
    k_cvt<<<(NPW + 255) / 256, 256, 0, stream>>>(sep5_pw2, w5p2, NPW);
    k_cvt<<<(NPW + 255) / 256, 256, 0, stream>>>(dil3_pw, wd3, NPW);
    k_cvt<<<(NPW + 255) / 256, 256, 0, stream>>>(dil5_pw, wd5, NPW);

    // 4) preprocess: state0/1 = BN(conv1x1(relu(s0/1)))  (K=512 WMMA GEMM)
    k_gemm<<<GGEMM, 256, 0, stream>>>(s0, 512L * HW, wpre0, st0, (long)CCH * HW,
                                      nullptr, 512, 1, 0);
    k_gemm<<<GGEMM, 256, 0, stream>>>(s1, 512L * HW, wpre1, st1, (long)CCH * HW,
                                      nullptr, 512, 1, 0);

    // 5) mixed-op DAG; states 2..5 live in d_out at channel base i*128
    const float* sptr[6] = { st0, st1,
                             out + 0 * CCH * HW, out + 1 * CCH * HW,
                             out + 2 * CCH * HW, out + 3 * CCH * HW };
    long sstr[6] = { (long)CCH * HW, (long)CCH * HW,
                     512L * HW, 512L * HW, 512L * HW, 512L * HW };

    int off = 0;
    for (int i = 0; i < 4; ++i) {
        float* dst = out + i * CCH * HW;
        const long dbs = 512L * HW;
        for (int j = 0; j < 2 + i; ++j) {
            const int m = off + j;
            const float* x = sptr[j];
            const long xbs = sstr[j];
            const float* pm = pG + m * 256;   // p[m][b][op] at pm[b*8+op]

            // max_pool / avg_pool / skip
            k_pool<<<GDW, 256, 0, stream>>>(x, xbs, dst, dbs, pm);

            // sep_conv_3x3
            k_dw<<<GDW, 256, 0, stream>>>(x, xbs, sep3_dw1 + m * CCH * 9, tmp1, 3, 1, 1);
            k_gemm<<<GGEMM, 256, 0, stream>>>(tmp1, (long)CCH * HW, w3p1 + m * 16384,
                                              tmp2, (long)CCH * HW, nullptr, CCH, 0, 0);
            k_dw<<<GDW, 256, 0, stream>>>(tmp2, (long)CCH * HW, sep3_dw2 + m * CCH * 9, tmp1, 3, 1, 1);
            k_gemm<<<GGEMM, 256, 0, stream>>>(tmp1, (long)CCH * HW, w3p2 + m * 16384,
                                              dst, dbs, pm + 4, CCH, 0, 1);

            // sep_conv_5x5
            k_dw<<<GDW, 256, 0, stream>>>(x, xbs, sep5_dw1 + m * CCH * 25, tmp1, 5, 2, 1);
            k_gemm<<<GGEMM, 256, 0, stream>>>(tmp1, (long)CCH * HW, w5p1 + m * 16384,
                                              tmp2, (long)CCH * HW, nullptr, CCH, 0, 0);
            k_dw<<<GDW, 256, 0, stream>>>(tmp2, (long)CCH * HW, sep5_dw2 + m * CCH * 25, tmp1, 5, 2, 1);
            k_gemm<<<GGEMM, 256, 0, stream>>>(tmp1, (long)CCH * HW, w5p2 + m * 16384,
                                              dst, dbs, pm + 5, CCH, 0, 1);

            // dil_conv_3x3 (dil=2, pad=2)
            k_dw<<<GDW, 256, 0, stream>>>(x, xbs, dil3_dw + m * CCH * 9, tmp1, 3, 2, 2);
            k_gemm<<<GGEMM, 256, 0, stream>>>(tmp1, (long)CCH * HW, wd3 + m * 16384,
                                              dst, dbs, pm + 6, CCH, 0, 1);

            // dil_conv_5x5 (dil=2, pad=4)
            k_dw<<<GDW, 256, 0, stream>>>(x, xbs, dil5_dw + m * CCH * 25, tmp1, 5, 4, 2);
            k_gemm<<<GGEMM, 256, 0, stream>>>(tmp1, (long)CCH * HW, wd5 + m * 16384,
                                              dst, dbs, pm + 7, CCH, 0, 1);
        }
        off += 2 + i;
    }
}
